// graph_geneset_attention_50620484550790
// MI455X (gfx1250) — matrix operand order repaired and checked by reading, not compiled, not run on hardware
//
#include <hip/hip_runtime.h>

typedef _Float16 v8h  __attribute__((ext_vector_type(8)));
typedef _Float16 v16h __attribute__((ext_vector_type(16)));
typedef float    v8f  __attribute__((ext_vector_type(8)));

#define NGS 64
constexpr int KP1   = 992;   // 978 padded up to multiple of 32
constexpr int BATCH = 1024;
constexpr int SRC   = 978;
constexpr int TOTAL = 18432; // sum of geneset sizes

struct Meta {
  int gs, sd, sdp, k2p;
  int outofs;   // column offset in target_feat / output
  int sdpofs;   // offset into packed b1 (floats), prefix of sdp
  int w1sofs;   // offset into swizzled W1 (halves)
  int w2sofs;   // offset into swizzled W2 (halves)
  int seofs;    // offset into SE buffer (halves)
};
struct MetaTab {
  Meta m[NGS];
  int w1s_total, w2s_total, sdp_total, out_total;
};
constexpr MetaTab make_meta() {
  MetaTab t = {};
  int outofs = 0, sdpofs = 0, w1 = 0, w2 = 0;
  const int bases[8] = {64, 128, 192, 256, 320, 384, 448, 512};
  for (int i = 0; i < NGS; ++i) {
    int gs  = bases[i % 8];
    int sd  = gs / 4 + 1;
    int sdp = ((sd + 31) / 32) * 32;   // pad se width to K-tile multiple
    int k2p = gs + sdp;                // padded concat-K (multiple of 32)
    t.m[i].gs = gs; t.m[i].sd = sd; t.m[i].sdp = sdp; t.m[i].k2p = k2p;
    t.m[i].outofs = outofs; t.m[i].sdpofs = sdpofs;
    t.m[i].w1sofs = w1; t.m[i].w2sofs = w2; t.m[i].seofs = sdpofs * BATCH;
    outofs += gs; sdpofs += sdp; w1 += sdp * KP1; w2 += gs * k2p;
  }
  t.w1s_total = w1; t.w2s_total = w2; t.sdp_total = sdpofs; t.out_total = outofs;
  return t;
}
constexpr MetaTab HM = make_meta();
static_assert(HM.out_total == TOTAL, "geneset column total mismatch");
static_assert(HM.sdp_total == 6144, "sdp total mismatch");
__constant__ MetaTab G = make_meta();

// ---- workspace layout (bytes) ----
constexpr size_t OFS_TSH = 0;                                          // t_source f16, 1024 x 992
constexpr size_t OFS_W1S = OFS_TSH + (size_t)BATCH * KP1 * 2;          // swizzled W1 f16
constexpr size_t OFS_SEH = OFS_W1S + (size_t)HM.w1s_total * 2;         // se f16, per-g 1024 x sdp
constexpr size_t OFS_W2S = OFS_SEH + (size_t)HM.sdp_total * BATCH * 2; // swizzled W2 f16
constexpr size_t OFS_B1  = OFS_W2S + (size_t)HM.w2s_total * 2;         // packed b1 f32 (padded)
constexpr size_t OFS_B2  = OFS_B1 + (size_t)HM.sdp_total * 4;          // packed b2 f32
constexpr size_t WS_NEED = OFS_B2 + (size_t)HM.out_total * 4;          // ~44.5 MB

struct ParamPtrs {
  const float* W1[NGS];
  const float* b1[NGS];
  const float* W2[NGS];
  const float* b2[NGS];
};

// ---------------- WMMA helpers ----------------
__device__ __forceinline__ v8f wmma_f16(v16h a, v16h b, v8f c) {
  return __builtin_amdgcn_wmma_f32_16x16x32_f16(false, a, false, b, (short)0, c,
                                                false, false);
}

// A fragment: 16x32 f16 tile from row-major [ld] buffer starting at column kb.
// ISA layout: lanes 0-15 hold M=lane, K(j) = j + 8*(j>=8); lanes 16-31 add K+8.
// => two contiguous 8-half (16B) runs per lane.
__device__ __forceinline__ v16h load_a_row(const _Float16* base, int ld, int kb,
                                           int lane) {
  int mrow = lane & 15, hi = lane >> 4;
  const _Float16* p = base + mrow * ld + kb + 8 * hi;
  v8h lo = *(const v8h*)p;
  v8h h8 = *(const v8h*)(p + 16);
  return __builtin_shufflevector(lo, h8, 0, 1, 2, 3, 4, 5, 6, 7, 8, 9, 10, 11,
                                 12, 13, 14, 15);
}

// B fragment from pre-swizzled weights: each 32x16 tile stored as 512
// contiguous halves, lane-major (lane*16 + j). Fully coalesced.
__device__ __forceinline__ v16h load_b_sw(const _Float16* tileBase, int lane) {
  const _Float16* p = tileBase + lane * 16;
  v8h lo = *(const v8h*)p;
  v8h h8 = *(const v8h*)(p + 8);
  return __builtin_shufflevector(lo, h8, 0, 1, 2, 3, 4, 5, 6, 7, 8, 9, 10, 11,
                                 12, 13, 14, 15);
}

// ---------------- kernel 1: convert t_source to padded f16 ----------------
__global__ void k_cvt_src(const float* __restrict__ src,
                          _Float16* __restrict__ dst) {
  int idx = blockIdx.x * 256 + threadIdx.x;
  if (idx >= BATCH * KP1) return;
  int r = idx / KP1, k = idx - r * KP1;
  dst[idx] = (k < SRC) ? (_Float16)src[r * SRC + k] : (_Float16)0.f;
}

// ---- kernel 2: convert + swizzle W1/W2 into WMMA-B layout, pack b1/b2 ----
__global__ void k_cvt_params(ParamPtrs P, char* __restrict__ ws) {
  int g = blockIdx.x;
  Meta m = G.m[g];
  _Float16* W1S = (_Float16*)(ws + OFS_W1S) + m.w1sofs;
  _Float16* W2S = (_Float16*)(ws + OFS_W2S) + m.w2sofs;
  float* B1 = (float*)(ws + OFS_B1) + m.sdpofs;
  float* B2 = (float*)(ws + OFS_B2) + m.outofs;

  int n1 = m.sdp * KP1;      // swizzled W1 halves (multiple of 512)
  int n2 = m.sdp;            // b1 floats
  int n3 = m.gs * m.k2p;     // swizzled W2 halves (multiple of 512)
  int n4 = m.gs;             // b2 floats
  int tot = n1 + n2 + n3 + n4;
  int stride = 256 * gridDim.y;
  for (int idx = threadIdx.x + blockIdx.y * 256; idx < tot; idx += stride) {
    if (idx < n1) {
      // W1 (978 x sd) -> swizzled (KP1 x sdp), zero padded
      int tile = idx >> 9, w = idx & 511;
      int lane = w >> 4, j = w & 15;
      const int kt = KP1 / 32;  // 31
      int ni = tile / kt, kb = (tile - ni * kt) * 32;
      int n = ni * 16 + (lane & 15);
      int k = kb + 8 * (lane >> 4) + j + 8 * (j >> 3);
      _Float16 v = (_Float16)0.f;
      if (n < m.sd && k < SRC) v = (_Float16)P.W1[g][k * m.sd + n];
      W1S[idx] = v;
    } else if (idx < n1 + n2) {
      int n = idx - n1;
      B1[n] = (n < m.sd) ? P.b1[g][n] : 0.f;
    } else if (idx < n1 + n2 + n3) {
      // W2 ((gs+sd) x gs) -> swizzled (k2p x gs), zero padded rows
      int i2 = idx - n1 - n2;
      int tile = i2 >> 9, w = i2 & 511;
      int lane = w >> 4, j = w & 15;
      int kt = m.k2p >> 5;
      int ni = tile / kt, kb = (tile - ni * kt) * 32;
      int n = ni * 16 + (lane & 15);
      int k = kb + 8 * (lane >> 4) + j + 8 * (j >> 3);
      _Float16 v = (_Float16)0.f;
      if (k < m.gs + m.sd) v = (_Float16)P.W2[g][k * m.gs + n];
      W2S[i2] = v;
    } else {
      int n = idx - n1 - n2 - n3;
      B2[n] = P.b2[g][n];
    }
  }
}

// ---- kernel 3: SE = relu(t_source @ W1 + b1), f16 out ----
// grid (64 genesets, 16 row-blocks of 64), 256 threads = 8 waves.
// Wave owns mi = wave&3, ni = (wave>>2) step 2; compile-time NA accumulators,
// one A fragment per K-step feeds NA back-to-back WMMAs.
template <int NA>
__device__ __forceinline__ void gemm1_body(const _Float16* __restrict__ arow,
                                           const _Float16* __restrict__ W1S,
                                           const float* __restrict__ B1,
                                           _Float16* __restrict__ SEH, int sdp,
                                           int aRow, int ni0, int lane) {
  const int kt = KP1 / 32;  // 31
  int hi = lane >> 4;
  v8f acc[NA];
#pragma unroll
  for (int u = 0; u < NA; ++u)
    acc[u] = v8f{0.f, 0.f, 0.f, 0.f, 0.f, 0.f, 0.f, 0.f};

  for (int kk = 0; kk < kt; ++kk) {
    v16h a = load_a_row(arow, KP1, kk * 32, lane);
#pragma unroll
    for (int u = 0; u < NA; ++u) {
      v16h b = load_b_sw(W1S + ((ni0 + 2 * u) * kt + kk) * 512, lane);
      acc[u] = wmma_f16(a, b, acc[u]);
    }
  }

#pragma unroll
  for (int u = 0; u < NA; ++u) {
    int col = (ni0 + 2 * u) * 16 + (lane & 15);
    float bias = B1[col];
#pragma unroll
    for (int i = 0; i < 8; ++i) {
      float v = fmaxf(acc[u][i] + bias, 0.f);
      SEH[(aRow + i + 8 * hi) * sdp + col] = (_Float16)v;
    }
  }
}

__global__ void k_gemm1(char* __restrict__ ws) {
  int g = blockIdx.x;
  Meta m = G.m[g];
  const _Float16* TSH = (const _Float16*)(ws + OFS_TSH);
  const _Float16* W1S = (const _Float16*)(ws + OFS_W1S) + m.w1sofs;
  const float* B1 = (const float*)(ws + OFS_B1) + m.sdpofs;
  _Float16* SEH = (_Float16*)(ws + OFS_SEH) + m.seofs;

  int wave = threadIdx.x >> 5, lane = threadIdx.x & 31;
  int mi = wave & 3, ni0 = wave >> 2;  // ni = ni0, ni0+2, ...
  int aRow = blockIdx.y * 64 + mi * 16;
  const _Float16* arow = TSH + aRow * KP1;

  // nT = sdp/16 is even (2..10); nAcc = nT/2 is block-uniform (1..5)
  int nAcc = m.sdp >> 5;
  switch (nAcc) {
    case 1: gemm1_body<1>(arow, W1S, B1, SEH, m.sdp, aRow, ni0, lane); break;
    case 2: gemm1_body<2>(arow, W1S, B1, SEH, m.sdp, aRow, ni0, lane); break;
    case 3: gemm1_body<3>(arow, W1S, B1, SEH, m.sdp, aRow, ni0, lane); break;
    case 4: gemm1_body<4>(arow, W1S, B1, SEH, m.sdp, aRow, ni0, lane); break;
    default: gemm1_body<5>(arow, W1S, B1, SEH, m.sdp, aRow, ni0, lane); break;
  }
}

// ---- kernel 4: logits = tanh([tgt|se] @ W2 + b2); softmax; *tgt ----
// grid (64 genesets, 64 row-blocks of 16), 256 threads = 8 waves, dyn LDS.
// Wave owns ni = wave, wave+8, ...; compile-time NA accumulators per wave.
template <int NA>
__device__ __forceinline__ void gemm2_body(const _Float16* __restrict__ As,
                                           int k2p, int gs,
                                           const _Float16* __restrict__ W2S,
                                           const float* __restrict__ B2,
                                           float* __restrict__ Ls, int kt,
                                           int wave, int lane) {
  int hi = lane >> 4;
  v8f acc[NA];
#pragma unroll
  for (int u = 0; u < NA; ++u)
    acc[u] = v8f{0.f, 0.f, 0.f, 0.f, 0.f, 0.f, 0.f, 0.f};

  for (int kk = 0; kk < kt; ++kk) {
    v16h a = load_a_row(As, k2p, kk * 32, lane);
#pragma unroll
    for (int u = 0; u < NA; ++u) {
      v16h b = load_b_sw(W2S + ((wave + 8 * u) * kt + kk) * 512, lane);
      acc[u] = wmma_f16(a, b, acc[u]);
    }
  }

#pragma unroll
  for (int u = 0; u < NA; ++u) {
    int col = (wave + 8 * u) * 16 + (lane & 15);
    float bias = B2[col];
#pragma unroll
    for (int i = 0; i < 8; ++i) {
      int r = i + 8 * hi;
      Ls[r * gs + col] = tanhf(acc[u][i] + bias);
    }
  }
}

__global__ void k_gemm2(const float* __restrict__ tf, char* __restrict__ ws,
                        float* __restrict__ out) {
  int g = blockIdx.x;
  Meta m = G.m[g];
  const _Float16* SEH = (const _Float16*)(ws + OFS_SEH) + m.seofs;
  const _Float16* W2S = (const _Float16*)(ws + OFS_W2S) + m.w2sofs;
  const float* B2 = (const float*)(ws + OFS_B2) + m.outofs;
  int rowBase = blockIdx.y * 16;

  extern __shared__ char smem[];
  _Float16* As = (_Float16*)smem;                    // 16 x k2p f16
  float* Ls = (float*)(smem + 16 * m.k2p * 2);       // 16 x gs f32 logits

  // stage concat A block: [ tgt(f32->f16) | se(f16, already zero padded) ]
  for (int r = 0; r < 16; ++r) {
    int grow = rowBase + r;
    const float* tgrow = tf + (size_t)grow * TOTAL + m.outofs;
    const _Float16* serow = SEH + grow * m.sdp;
    for (int c = threadIdx.x; c < m.k2p; c += 256) {
      As[r * m.k2p + c] =
          (c < m.gs) ? (_Float16)tgrow[c] : serow[c - m.gs];
    }
  }
  __syncthreads();

  int wave = threadIdx.x >> 5, lane = threadIdx.x & 31;
  int nT = m.gs >> 4;        // 4..32 N tiles
  int kt = m.k2p >> 5;       // 3..21 K steps
  int nAcc = (nT - wave + 7) >> 3;  // 0..4, wave-uniform
  switch (nAcc) {
    case 1: gemm2_body<1>(As, m.k2p, m.gs, W2S, B2, Ls, kt, wave, lane); break;
    case 2: gemm2_body<2>(As, m.k2p, m.gs, W2S, B2, Ls, kt, wave, lane); break;
    case 3: gemm2_body<3>(As, m.k2p, m.gs, W2S, B2, Ls, kt, wave, lane); break;
    case 4: gemm2_body<4>(As, m.k2p, m.gs, W2S, B2, Ls, kt, wave, lane); break;
    default: break;  // nAcc == 0: no tiles for this wave
  }
  __syncthreads();

  // row softmax over gs, scale by fp32 tgt; 2 rows per wave
  for (int r = wave * 2; r < wave * 2 + 2; ++r) {
    float* row = Ls + r * m.gs;
    float mx = -3.0e38f;
    for (int c = lane; c < m.gs; c += 32) mx = fmaxf(mx, row[c]);
    for (int o = 16; o > 0; o >>= 1) mx = fmaxf(mx, __shfl_xor(mx, o, 32));
    float s = 0.f;
    for (int c = lane; c < m.gs; c += 32) {
      float e = __expf(row[c] - mx);
      row[c] = e;
      s += e;
    }
    for (int o = 16; o > 0; o >>= 1) s += __shfl_xor(s, o, 32);
    float inv = 1.0f / s;
    int grow = rowBase + r;
    const float* tgrow = tf + (size_t)grow * TOTAL + m.outofs;
    float* orow = out + (size_t)grow * TOTAL + m.outofs;
    for (int c = lane; c < m.gs; c += 32) orow[c] = row[c] * inv * tgrow[c];
  }
}

extern "C" void kernel_launch(void* const* d_in, const int* in_sizes, int n_in,
                              void* d_out, int out_size, void* d_ws,
                              size_t ws_size, hipStream_t stream) {
  (void)in_sizes; (void)n_in; (void)out_size;
  if (ws_size < WS_NEED) return;  // workspace layout needs ~44.5 MB

  const float* target_feat = (const float*)d_in[0];
  const float* t_source = (const float*)d_in[1];
  ParamPtrs P;
  for (int g = 0; g < NGS; ++g) {
    P.W1[g] = (const float*)d_in[2 + 4 * g + 0];
    P.b1[g] = (const float*)d_in[2 + 4 * g + 1];
    P.W2[g] = (const float*)d_in[2 + 4 * g + 2];
    P.b2[g] = (const float*)d_in[2 + 4 * g + 3];
  }
  char* ws = (char*)d_ws;

  k_cvt_src<<<(BATCH * KP1 + 255) / 256, 256, 0, stream>>>(
      t_source, (_Float16*)(ws + OFS_TSH));
  k_cvt_params<<<dim3(NGS, 32), 256, 0, stream>>>(P, ws);
  k_gemm1<<<dim3(NGS, 16), 256, 0, stream>>>(ws);
  constexpr int SMEM2 = 16 * 672 * 2 + 16 * 512 * 4;  // 54,272 B
  k_gemm2<<<dim3(NGS, 64), 256, SMEM2, stream>>>(target_feat, ws,
                                                 (float*)d_out);
}
